// SelfAttention_77781857731050
// MI455X (gfx1250) — matrix operand order repaired and checked by reading, not compiled, not run on hardware
//
#include <hip/hip_runtime.h>
#include <hip/hip_bf16.h>
#include <stdint.h>

// ---------- types ----------
typedef __attribute__((ext_vector_type(16))) __bf16 v16bf;
typedef __attribute__((ext_vector_type(8)))  float  v8f;

union FragAB { v16bf v; uint32_t d[8]; };   // 16x32 (A) or 32x16 (B) bf16 fragment, 8 VGPRs

// TDM descriptor vector types (6-arg builtin signature: clang-23 / therock-10.0)
typedef __attribute__((ext_vector_type(4))) unsigned int tdm_v4u;
typedef __attribute__((ext_vector_type(8))) int          tdm_v8i;
typedef __attribute__((ext_vector_type(4))) int          tdm_v4i;

#if defined(__has_builtin)
#if __has_builtin(__builtin_amdgcn_tensor_load_to_lds) && __has_builtin(__builtin_amdgcn_s_wait_tensorcnt)
#define ATHENA_USE_TDM 1
#endif
#endif

#define HD      128
#define NH      16
#define SEQ     2048
#define DMODEL  2048
#define BATCH   2
#define SCALE_QK 0.08838834764831845f   // 1/sqrt(128)
#define LN10000  9.210340371976184f

// ---------- small helpers ----------
static __device__ inline unsigned short f2bf(float f) {
    unsigned int u = __float_as_uint(f);
    u += 0x7FFFu + ((u >> 16) & 1u);           // round-to-nearest-even
    return (unsigned short)(u >> 16);
}
static __device__ inline float bf2f(unsigned short u) {
    return __uint_as_float(((unsigned int)u) << 16);
}
static __device__ inline float rowmax16(float v) {
    #pragma unroll
    for (int m = 1; m < 16; m <<= 1) v = fmaxf(v, __shfl_xor(v, m, 16));
    return v;
}
static __device__ inline float rowsum16(float v) {
    #pragma unroll
    for (int m = 1; m < 16; m <<= 1) v += __shfl_xor(v, m, 16);
    return v;
}

#ifdef ATHENA_USE_TDM
// Issue a 1-D TDM DMA: nbytes (multiple of 8, tile_dim0 <= 65535 elems) global -> LDS.
// D# layout per CDNA5 ISA.  6-arg builtin:
// (v4u group0, v8i group1, v4i group2, v4i group3, v8i group4(reserved), i32 cpol)
static __device__ inline void tdm_load_1d(unsigned int lds_addr, const void* gptr, int nbytes) {
    const unsigned long long ga = (unsigned long long)(size_t)gptr;
    const int elems = nbytes >> 3;                  // 8-byte elements
    tdm_v4u g0;
    g0[0] = 1u;                                      // count=1, user mode
    g0[1] = lds_addr;                                // LDS byte address
    g0[2] = (unsigned int)ga;                        // global_addr[31:0]
    g0[3] = (unsigned int)((ga >> 32) & 0x1FFFFFFu)  // global_addr[56:32]
          | (2u << 30);                              // type=2 ("image")
    tdm_v8i g1;
    g1[0] = 0x00030000;                              // workgroup_mask=0, data_size=3 (8B)
    g1[1] = (int)((unsigned)(elems & 0xFFFF) << 16); // tensor_dim0[15:0]
    g1[2] = (int)(((unsigned)(elems >> 16) & 0xFFFFu) | (1u << 16)); // dim0 hi | tensor_dim1=1
    g1[3] = (int)((unsigned)(elems & 0xFFFF) << 16); // tile_dim0 = elems
    g1[4] = 0;                                       // tile_dim1=0, tile_dim2=0
    g1[5] = elems;                                   // tensor_dim0_stride[31:0]
    g1[6] = 0;
    g1[7] = 0;
    tdm_v4i z4 = {0, 0, 0, 0};
    tdm_v8i z8 = {0, 0, 0, 0, 0, 0, 0, 0};
    __builtin_amdgcn_tensor_load_to_lds(g0, g1, z4, z4, z8, 0);
}
#endif

// ---------- stage 0: fp32 -> bf16 ----------
__global__ __launch_bounds__(256) void cvt_f32_bf16(const float* __restrict__ src,
                                                    unsigned short* __restrict__ dst, int n) {
    for (int i = blockIdx.x * 256 + threadIdx.x; i < n; i += gridDim.x * 256)
        dst[i] = f2bf(src[i]);
}

// ---------- stages 1 & 4: bf16 WMMA GEMM, C = A * B^T + bias ----------
// A: MxK row-major bf16, Bw: NxK row-major bf16, K multiple of 32.
// Block tile 128(M) x 128(N); 8 waves in 4x2, each wave computes 32x64.
// MODE 0: write fp32 C (final projection).  MODE 1: scatter qkv into Q/K/V (B,H,S,hd) bf16.
template <int MODE>
__global__ __launch_bounds__(256) void gemm_bf16_wmma(
    const unsigned short* __restrict__ A,
    const unsigned short* __restrict__ Bw,
    const float* __restrict__ bias,
    float* __restrict__ Cf,
    unsigned short* __restrict__ Qb,
    unsigned short* __restrict__ Kb,
    unsigned short* __restrict__ Vb,
    int N, int K) {

    __shared__ uint32_t sA[128][16];   // 128 rows x 32 bf16 (16 dwords)
    __shared__ uint32_t sB[128][16];

    const int tid  = threadIdx.x;
    const int w    = tid >> 5;
    const int lane = tid & 31;
    const int h    = lane >> 4;
    const int ln   = lane & 15;
    const int m0   = blockIdx.y * 128;
    const int n0   = blockIdx.x * 128;
    const int wm   = (w >> 1) * 32;    // wave M offset in block tile
    const int wn   = (w & 1) * 64;     // wave N offset in block tile
    const int Kd   = K >> 1;           // dwords per row

    const uint32_t* Ad = (const uint32_t*)A;
    const uint32_t* Bd = (const uint32_t*)Bw;

    v8f acc[2][4];
    #pragma unroll
    for (int a = 0; a < 2; ++a)
        #pragma unroll
        for (int j = 0; j < 4; ++j) acc[a][j] = (v8f){};

    for (int k0 = 0; k0 < Kd; k0 += 16) {
        __syncthreads();
        for (int idx = tid; idx < 128 * 16; idx += 256) {
            int r = idx >> 4, c = idx & 15;
            sA[r][c] = Ad[(size_t)(m0 + r) * Kd + k0 + c];
            sB[r][c] = Bd[(size_t)(n0 + r) * Kd + k0 + c];
        }
        // prefetch next A tile slice into GL2 while this K-step computes
        if (k0 + 16 < Kd)
            __builtin_prefetch(&Ad[(size_t)(m0 + (tid >> 1)) * Kd + k0 + 16 + 8 * (tid & 1)], 0, 1);
        __syncthreads();

        FragAB fa[2];
        #pragma unroll
        for (int a = 0; a < 2; ++a) {
            const int ar = wm + a * 16 + ln;
            #pragma unroll
            for (int v = 0; v < 8; ++v)
                fa[a].d[v] = sA[ar][8 * h + v];
        }
        #pragma unroll
        for (int j = 0; j < 4; ++j) {
            FragAB fb;
            #pragma unroll
            for (int v = 0; v < 8; ++v)
                fb.d[v] = sB[wn + j * 16 + ln][8 * h + v];
            #pragma unroll
            for (int a = 0; a < 2; ++a)
                acc[a][j] = __builtin_amdgcn_wmma_f32_16x16x32_bf16(
                    false, fa[a].v, false, fb.v, (short)0, acc[a][j], false, false);
        }
    }

    #pragma unroll
    for (int a = 0; a < 2; ++a)
        #pragma unroll
        for (int j = 0; j < 4; ++j)
            #pragma unroll
            for (int i = 0; i < 8; ++i) {
                const int row = m0 + wm + a * 16 + i + 8 * h;
                const int col = n0 + wn + j * 16 + ln;
                const float val = acc[a][j][i] + bias[col];
                if constexpr (MODE == 0) {
                    Cf[(size_t)row * N + col] = val;
                } else {
                    const int head = col / 384;
                    const int jj   = col - head * 384;  // 0..127 q, 128..255 k, 256..383 v
                    const int b    = row >> 11;         // /SEQ
                    const int s    = row & (SEQ - 1);
                    const size_t base = (((size_t)(b * NH + head)) * SEQ + s) * HD;
                    const unsigned short bv = f2bf(val);
                    if (jj < 128)       Qb[base + jj]        = bv;
                    else if (jj < 256)  Kb[base + jj - 128]  = bv;
                    else                Vb[base + jj - 256]  = bv;
                }
            }
}

// ---------- stage 2: RoPE in place on (B,H,S,hd) bf16, first 32 dims ----------
__global__ __launch_bounds__(256) void rope_inplace(unsigned short* __restrict__ buf) {
    const int t = blockIdx.x * 256 + threadIdx.x;      // BATCH*NH*SEQ*16 threads
    if (t >= BATCH * NH * SEQ * 16) return;
    const int p  = t & 15;
    const int s  = (t >> 4) & (SEQ - 1);
    const int bh = t >> 15;
    const size_t base = ((size_t)bh * SEQ + s) * HD;
    const float x1 = bf2f(buf[base + p]);
    const float x2 = bf2f(buf[base + p + 16]);
    const float invf = __expf(-((float)(2 * p) / 32.0f) * LN10000);
    const float th = (float)s * invf;
    float sn, cs;
    __sincosf(th, &sn, &cs);
    buf[base + p]      = f2bf(x1 * cs - x2 * sn);
    buf[base + p + 16] = f2bf(x2 * cs + x1 * sn);
}

// ---------- stage 3: causal flash attention, bf16 WMMA, fp32 accum ----------
// grid: (SEQ/128, BATCH*NH), block 256 = 8 waves; wave w owns q rows [q0+16w, q0+16w+16)
__global__ __launch_bounds__(256) void attn_flash_wmma(
    const unsigned short* __restrict__ Qb,
    const unsigned short* __restrict__ Kb,
    const unsigned short* __restrict__ Vb,
    unsigned short* __restrict__ Ctx) {

    __shared__ uint32_t kT[32][64];        // 32 keys x 128 dims bf16 (8 KB)
    __shared__ uint32_t vT[32][64];
    __shared__ uint32_t pTd[8][16][16];    // per-wave 16x32 bf16 P tile

    const int tid  = threadIdx.x;
    const int w    = tid >> 5;
    const int lane = tid & 31;
    const int h    = lane >> 4;
    const int ln   = lane & 15;
    const int bh   = blockIdx.y;
    const int b    = bh >> 4;
    const int hh   = bh & 15;
    const int q0   = blockIdx.x * 128;
    const int qw   = q0 + w * 16;

    const uint32_t* Qd = (const uint32_t*)Qb;   // 64 dwords per (token,head) row
#ifndef ATHENA_USE_TDM
    const uint32_t* Kdw = (const uint32_t*)Kb;
    const uint32_t* Vdw = (const uint32_t*)Vb;
#endif

    // Q fragments: 16 rows x 128 dims = 4 A-fragments (K=32 each)
    FragAB qf[4];
    {
        const size_t rb = ((size_t)bh * SEQ + (qw + ln)) * 64;
        #pragma unroll
        for (int c = 0; c < 4; ++c)
            #pragma unroll
            for (int v = 0; v < 8; ++v)
                qf[c].d[v] = Qd[rb + c * 16 + 8 * h + v];
    }

    v8f o[8];
    float mrun[8], lrun[8];
    #pragma unroll
    for (int j = 0; j < 8; ++j) o[j] = (v8f){};
    #pragma unroll
    for (int i = 0; i < 8; ++i) { mrun[i] = -1e30f; lrun[i] = 0.0f; }

    const int nchunks = (q0 + 128) >> 5;
    for (int kc = 0; kc < nchunks; ++kc) {
        const int kb = kc << 5;
        __syncthreads();
#ifdef ATHENA_USE_TDM
        // Tensor Data Mover: one DMA each for the 32x128 bf16 K and V tiles (8 KB each).
        if (w == 0) {
            const size_t gofs = ((size_t)bh * SEQ + kb) * HD;   // elements
            tdm_load_1d((unsigned int)(size_t)&kT[0][0], Kb + gofs, 32 * HD * 2);
            tdm_load_1d((unsigned int)(size_t)&vT[0][0], Vb + gofs, 32 * HD * 2);
            __builtin_amdgcn_s_wait_tensorcnt(0);
        }
#else
        for (int idx = tid; idx < 32 * 64; idx += 256) {
            int r = idx >> 6, c = idx & 63;
            kT[r][c] = Kdw[((size_t)bh * SEQ + kb + r) * 64 + c];
            vT[r][c] = Vdw[((size_t)bh * SEQ + kb + r) * 64 + c];
        }
#endif
        __syncthreads();

        // scores: 16 q-rows x 32 keys as two 16x16 C tiles
        v8f s0 = (v8f){}, s1 = (v8f){};
        #pragma unroll
        for (int c = 0; c < 4; ++c) {
            FragAB kf0, kf1;
            #pragma unroll
            for (int v = 0; v < 8; ++v) {
                kf0.d[v] = kT[ln][16 * c + 8 * h + v];
                kf1.d[v] = kT[16 + ln][16 * c + 8 * h + v];
            }
            s0 = __builtin_amdgcn_wmma_f32_16x16x32_bf16(false, qf[c].v, false, kf0.v, (short)0, s0, false, false);
            s1 = __builtin_amdgcn_wmma_f32_16x16x32_bf16(false, qf[c].v, false, kf1.v, (short)0, s1, false, false);
        }

        // online softmax (row = i + 8h across 16 lanes, col = lane%16)
        unsigned short* pU = (unsigned short*)pTd;
        #pragma unroll
        for (int i = 0; i < 8; ++i) {
            const int row = qw + i + 8 * h;
            const int c0 = kb + ln, c1 = c0 + 16;
            float a0 = (c0 <= row) ? s0[i] * SCALE_QK : -10000.0f;
            float a1 = (c1 <= row) ? s1[i] * SCALE_QK : -10000.0f;
            const float mx   = rowmax16(fmaxf(a0, a1));
            const float mnew = fmaxf(mrun[i], mx);
            const float alpha = __expf(mrun[i] - mnew);
            const float p0 = __expf(a0 - mnew);
            const float p1 = __expf(a1 - mnew);
            const float rs = rowsum16(p0 + p1);
            lrun[i] = lrun[i] * alpha + rs;
            mrun[i] = mnew;
            #pragma unroll
            for (int j = 0; j < 8; ++j) o[j][i] *= alpha;
            const int rl = i + 8 * h;
            pU[(w * 16 + rl) * 32 + ln]      = f2bf(p0);
            pU[(w * 16 + rl) * 32 + 16 + ln] = f2bf(p1);
        }

        // P as A-fragment (wave-private LDS transpose; same-wave DS ops are in order)
        FragAB pf;
        const uint32_t* pR = &pTd[w][0][0];
        #pragma unroll
        for (int v = 0; v < 8; ++v)
            pf.d[v] = pR[ln * 16 + 8 * h + v];

        // PV: 8 B-fragments over the 128 output dims
        const unsigned short* vU = (const unsigned short*)vT;
        #pragma unroll
        for (int j = 0; j < 8; ++j) {
            FragAB vf;
            #pragma unroll
            for (int v = 0; v < 8; ++v) {
                const int k0 = 16 * h + 2 * v;
                const unsigned int lo = vU[k0 * 128 + j * 16 + ln];
                const unsigned int hi = vU[(k0 + 1) * 128 + j * 16 + ln];
                vf.d[v] = lo | (hi << 16);
            }
            o[j] = __builtin_amdgcn_wmma_f32_16x16x32_bf16(false, pf.v, false, vf.v, (short)0, o[j], false, false);
        }
    }

    // normalize and write ctx as (B, S, D) bf16 with col = head*128 + dim
    #pragma unroll
    for (int j = 0; j < 8; ++j)
        #pragma unroll
        for (int i = 0; i < 8; ++i) {
            const int row = qw + i + 8 * h;
            const float val = o[j][i] * (1.0f / lrun[i]);
            Ctx[((size_t)(b * SEQ + row)) * DMODEL + hh * HD + j * 16 + ln] = f2bf(val);
        }
}

// ---------- host-side launch ----------
extern "C" void kernel_launch(void* const* d_in, const int* in_sizes, int n_in,
                              void* d_out, int out_size, void* d_ws, size_t ws_size,
                              hipStream_t stream) {
    const float* x     = (const float*)d_in[0];   // (2,2048,2048)
    const float* W_qkv = (const float*)d_in[1];   // (6144,2048)
    const float* b_qkv = (const float*)d_in[2];   // (6144,)
    const float* W_out = (const float*)d_in[3];   // (2048,2048)
    const float* b_out = (const float*)d_in[4];   // (2048,)
    float* out = (float*)d_out;

    const size_t M  = (size_t)BATCH * SEQ;        // 4096
    const size_t nX    = M * DMODEL;              // 8388608
    const size_t nWqkv = (size_t)3 * DMODEL * DMODEL;
    const size_t nWout = (size_t)DMODEL * DMODEL;
    const size_t nQKV  = (size_t)BATCH * NH * SEQ * HD;  // per Q/K/V

    char* ws = (char*)d_ws;
    size_t off = 0;
    unsigned short* Xb    = (unsigned short*)(ws + off); off += nX    * 2;
    unsigned short* Wqb   = (unsigned short*)(ws + off); off += nWqkv * 2;
    unsigned short* Wob   = (unsigned short*)(ws + off); off += nWout * 2;
    unsigned short* Qb    = (unsigned short*)(ws + off); off += nQKV  * 2;
    unsigned short* Kb    = (unsigned short*)(ws + off); off += nQKV  * 2;
    unsigned short* Vb    = (unsigned short*)(ws + off); off += nQKV  * 2;
    unsigned short* Ctxb  = (unsigned short*)(ws + off); off += nX    * 2;

    // stage 0: convert operands to bf16
    cvt_f32_bf16<<<dim3(2048), dim3(256), 0, stream>>>(x,     Xb,  (int)nX);
    cvt_f32_bf16<<<dim3(2048), dim3(256), 0, stream>>>(W_qkv, Wqb, (int)nWqkv);
    cvt_f32_bf16<<<dim3(2048), dim3(256), 0, stream>>>(W_out, Wob, (int)nWout);

    // stage 1: qkv projection, scatter to Q/K/V (B,H,S,hd)
    gemm_bf16_wmma<1><<<dim3(3 * DMODEL / 128, M / 128), dim3(256), 0, stream>>>(
        Xb, Wqb, b_qkv, nullptr, Qb, Kb, Vb, 3 * DMODEL, DMODEL);

    // stage 2: RoPE on q and k
    const int ropeThreads = BATCH * NH * SEQ * 16;
    rope_inplace<<<dim3(ropeThreads / 256), dim3(256), 0, stream>>>(Qb);
    rope_inplace<<<dim3(ropeThreads / 256), dim3(256), 0, stream>>>(Kb);

    // stage 3: causal attention
    attn_flash_wmma<<<dim3(SEQ / 128, BATCH * NH), dim3(256), 0, stream>>>(Qb, Kb, Vb, Ctxb);

    // stage 4: output projection -> fp32 d_out
    gemm_bf16_wmma<0><<<dim3(DMODEL / 128, M / 128), dim3(256), 0, stream>>>(
        Ctxb, Wob, b_out, out, nullptr, nullptr, nullptr, DMODEL, DMODEL);

    (void)in_sizes; (void)n_in; (void)out_size; (void)ws_size;
}
